// PointNetSOP_71863392797204
// MI455X (gfx1250) — matrix operand order repaired and checked by reading, not compiled, not run on hardware
//
#include <hip/hip_runtime.h>

// ---------------------------------------------------------------------------
// PointNet + second-order pooling (covariance sqrt) for MI455X / gfx1250.
//  - GEMMs: v_wmma_f32_16x16x32_bf16, wave32, 16x64 output per wave
//    (A-fragment reused across 4 WMMAs -> 4x less L2 traffic, deep WMMA chain)
//  - Matrix sqrt: Newton-Schulz with split-bf16 (bf16x3) ~f32 GEMMs, also
//    4x J-blocked so the A-side hi/lo split is amortized over 12 WMMAs/K-step
//  - FC: HBM-bound; Wfc streamed through LDS with async double buffering
//    (global_load_async_to_lds_b128 + s_wait_asynccnt)
// ---------------------------------------------------------------------------

typedef unsigned short u16;
typedef unsigned int   u32;
typedef __attribute__((ext_vector_type(16))) __bf16 v16bf;
typedef __attribute__((ext_vector_type(8)))  float  v8f;
typedef __attribute__((ext_vector_type(8)))  u16    v8u;
typedef __attribute__((ext_vector_type(16))) u16    v16u;

#define NB       16
#define NPTS     10000
#define MROWS    (NB * NPTS)          // 160000 rows (points)
#define FEAT     256
#define FEAT2    (FEAT * FEAT)
#define OUTD     256
#define BN_EPS   1e-5f
#define NS_ITERS 14
#define ROWCHUNK 256
#define NCHUNK   (MROWS / ROWCHUNK)   // 625
#define FC_CH    1024                 // floats per LDS stage buffer

// ---- bf16 helpers (round-to-nearest-even) ----
__device__ __forceinline__ u16 f2bf(float f) {
  u32 u = __builtin_bit_cast(u32, f);
  u32 r = u + 0x7FFFu + ((u >> 16) & 1u);
  return (u16)(r >> 16);
}
__device__ __forceinline__ float bf2f(u16 h) {
  return __builtin_bit_cast(float, (u32)h << 16);
}

__device__ __forceinline__ v8f wmma_bf16(const v16u& a, const v16u& b, v8f c) {
  return __builtin_amdgcn_wmma_f32_16x16x32_bf16(
      /*neg_a=*/false, __builtin_bit_cast(v16bf, a),
      /*neg_b=*/false, __builtin_bit_cast(v16bf, b),
      /*c_mod=*/(short)0, c, /*reuse_a=*/false, /*reuse_b=*/false);
}

// ---------------------------------------------------------------------------
// Prep kernels
// ---------------------------------------------------------------------------

__global__ void k_prep_x0(const float* __restrict__ pts, u16* __restrict__ x0) {
  int r = blockIdx.x * blockDim.x + threadIdx.x;
  if (r >= MROWS) return;
  float p0 = pts[r * 3 + 0], p1 = pts[r * 3 + 1], p2 = pts[r * 3 + 2];
  u16* d = x0 + (size_t)r * 32;
  d[0] = f2bf(p0); d[1] = f2bf(p1); d[2] = f2bf(p2);
  #pragma unroll
  for (int c = 3; c < 32; ++c) d[c] = 0;
}

__global__ void k_pad_w0(const float* __restrict__ w, u16* __restrict__ wb) {
  int t = blockIdx.x * blockDim.x + threadIdx.x;
  if (t >= 64 * 32) return;
  int r = t >> 5, c = t & 31;
  wb[t] = (c < 3) ? f2bf(w[r * 3 + c]) : (u16)0;
}

__global__ void k_cvt_bf16(const float* __restrict__ s, u16* __restrict__ d, int n) {
  int t = blockIdx.x * blockDim.x + threadIdx.x;
  if (t < n) d[t] = f2bf(s[t]);
}

// ---------------------------------------------------------------------------
// Activation GEMM: Y[M,N] = A[M,K](bf16) @ W[N,K]^T(bf16); one wave computes a
// 16x64 output strip: A fragment loaded once per K-step, reused by 4 WMMAs.
// ---------------------------------------------------------------------------
__global__ void k_gemm_act4(const u16* __restrict__ A, int lda,
                            const u16* __restrict__ W, int ldw,
                            u16* __restrict__ Y, int ldy, int K) {
  const int lane = threadIdx.x;
  const int mt = blockIdx.x;
  const int nbase = blockIdx.y * 64;
  const int half = lane >> 4, l15 = lane & 15;
  const int row = mt * 16 + l15;
  const int kbA = half * 16, kbB = half * 16;
  v8f acc[4] = {{}, {}, {}, {}};
  for (int k0 = 0; k0 < K; k0 += 32) {
    v8u a0 = *(const v8u*)(A + (size_t)row * lda + k0 + kbA);
    v8u a1 = *(const v8u*)(A + (size_t)row * lda + k0 + kbA + 8);
    v16u au;
    #pragma unroll
    for (int i = 0; i < 8; ++i) { au[i] = a0[i]; au[8 + i] = a1[i]; }
    #pragma unroll
    for (int j = 0; j < 4; ++j) {
      const int wrow = nbase + j * 16 + l15;
      v8u b0 = *(const v8u*)(W + (size_t)wrow * ldw + k0 + kbB);
      v8u b1 = *(const v8u*)(W + (size_t)wrow * ldw + k0 + kbB + 8);
      v16u bu;
      #pragma unroll
      for (int i = 0; i < 8; ++i) { bu[i] = b0[i]; bu[8 + i] = b1[i]; }
      acc[j] = wmma_bf16(au, bu, acc[j]);
    }
  }
  #pragma unroll
  for (int j = 0; j < 4; ++j) {
    #pragma unroll
    for (int i = 0; i < 8; ++i) {
      int r = mt * 16 + half * 8 + i;
      int c = nbase + j * 16 + l15;
      Y[(size_t)r * ldy + c] = f2bf(acc[j][i]);
    }
  }
}

// ---------------------------------------------------------------------------
// Deterministic column statistics (sum, sumsq) in two passes
// ---------------------------------------------------------------------------
__global__ void k_colstats(const u16* __restrict__ Y, int C,
                           float* __restrict__ psum, float* __restrict__ psumsq) {
  int c = threadIdx.x;               // blockDim.x == C
  int r0 = blockIdx.x * ROWCHUNK;
  float s = 0.f, s2 = 0.f;
  for (int i = 0; i < ROWCHUNK; ++i) {
    int r = r0 + i;
    if (r < MROWS) {
      float v = bf2f(Y[(size_t)r * C + c]);
      s += v; s2 += v * v;
    }
  }
  psum[(size_t)blockIdx.x * C + c] = s;
  psumsq[(size_t)blockIdx.x * C + c] = s2;
}

__global__ void k_colred(const float* __restrict__ psum, const float* __restrict__ psumsq,
                         int C, int nblk, float* __restrict__ sum, float* __restrict__ sumsq) {
  int c = threadIdx.x;
  float s = 0.f, s2 = 0.f;
  for (int i = 0; i < nblk; ++i) { s += psum[(size_t)i * C + c]; s2 += psumsq[(size_t)i * C + c]; }
  sum[c] = s; sumsq[c] = s2;
}

__global__ void k_bnfin(const float* __restrict__ sum, const float* __restrict__ sumsq,
                        const float* __restrict__ g, const float* __restrict__ b,
                        float* __restrict__ scale, float* __restrict__ shift) {
  int c = threadIdx.x;
  const float inv = 1.0f / (float)MROWS;
  float mu = sum[c] * inv;
  float var = sumsq[c] * inv - mu * mu;
  float sc = g[c] * rsqrtf(var + BN_EPS);
  scale[c] = sc;
  shift[c] = b[c] - mu * sc;
}

__global__ void k_bn_apply(const u16* __restrict__ Y, u16* __restrict__ X,
                           const float* __restrict__ scale, const float* __restrict__ shift,
                           int C, int total) {
  int stride = gridDim.x * blockDim.x;
  for (int i = blockIdx.x * blockDim.x + threadIdx.x; i < total; i += stride) {
    int c = i % C;
    float v = bf2f(Y[i]) * scale[c] + shift[c];
    X[i] = f2bf(fmaxf(v, 0.0f));
  }
}

__global__ void k_center(const u16* __restrict__ F, const float* __restrict__ fsum,
                         u16* __restrict__ Fc, int total) {
  const float inv = 1.0f / (float)MROWS;
  int stride = gridDim.x * blockDim.x;
  for (int i = blockIdx.x * blockDim.x + threadIdx.x; i < total; i += stride) {
    int c = i % FEAT;
    Fc[i] = f2bf(bf2f(F[i]) - fsum[c] * inv);
  }
}

// ---------------------------------------------------------------------------
// Covariance: Cov[b] = Fc[b]^T Fc[b] / N. One wave computes a 16x64 strip:
// the Fc^T A-fragment (lane-coalesced channel-contiguous loads) is reused by
// 4 WMMAs over 4 J-tiles.
// ---------------------------------------------------------------------------
__global__ void k_cov4(const u16* __restrict__ FcAll, float* __restrict__ Cov) {
  const int lane = threadIdx.x;
  const int it = blockIdx.x, jq = blockIdx.y, b = blockIdx.z;
  const int half = lane >> 4, l15 = lane & 15;
  const u16* F = FcAll + (size_t)b * NPTS * FEAT;
  const int ci = it * 16 + l15;
  const int cjbase = jq * 64;
  v8f acc[4] = {{}, {}, {}, {}};
  for (int n0 = 0; n0 < NPTS; n0 += 32) {
    v16u au;
    #pragma unroll
    for (int j = 0; j < 8; ++j) {
      int n1 = n0 + half * 16 + j;
      int n2 = n0 + half * 16 + 8 + j;
      au[j]     = (n1 < NPTS) ? F[(size_t)n1 * FEAT + ci] : (u16)0;
      au[8 + j] = (n2 < NPTS) ? F[(size_t)n2 * FEAT + ci] : (u16)0;
    }
    #pragma unroll
    for (int jt = 0; jt < 4; ++jt) {
      const int cj = cjbase + jt * 16 + l15;
      v16u bu;
      #pragma unroll
      for (int j = 0; j < 16; ++j) {
        int n = n0 + half * 16 + j;
        bu[j] = (n < NPTS) ? F[(size_t)n * FEAT + cj] : (u16)0;
      }
      acc[jt] = wmma_bf16(au, bu, acc[jt]);
    }
  }
  const float invN = 1.0f / (float)NPTS;
  #pragma unroll
  for (int jt = 0; jt < 4; ++jt) {
    #pragma unroll
    for (int i = 0; i < 8; ++i) {
      int r = it * 16 + half * 8 + i;
      int c = cjbase + jt * 16 + l15;
      Cov[(size_t)b * FEAT2 + (size_t)r * FEAT + c] = acc[jt][i] * invN;
    }
  }
}

// Frobenius norm per batch -> 1/s and sqrt(s)
__global__ void k_frob(const float* __restrict__ Cov, float* __restrict__ rs,
                       float* __restrict__ ss) {
  __shared__ float red[256];
  int b = blockIdx.x, t = threadIdx.x;
  const float* M = Cov + (size_t)b * FEAT2;
  float s = 0.f;
  for (int i = t; i < FEAT2; i += 256) { float v = M[i]; s += v * v; }
  red[t] = s; __syncthreads();
  for (int w = 128; w > 0; w >>= 1) { if (t < w) red[t] += red[t + w]; __syncthreads(); }
  if (t == 0) {
    float fr = sqrtf(red[0]) + 1e-30f;
    rs[b] = 1.0f / fr;
    ss[b] = sqrtf(fr);
  }
}

__global__ void k_ns_init(const float* __restrict__ Cov, const float* __restrict__ rs,
                          float* __restrict__ Y, float* __restrict__ Z) {
  int idx = blockIdx.x * blockDim.x + threadIdx.x;   // < NB*FEAT2
  int b = idx >> 16;
  int e = idx & (FEAT2 - 1);
  Y[idx] = Cov[idx] * rs[b];
  Z[idx] = ((e >> 8) == (e & 255)) ? 1.0f : 0.0f;
}

// ---------------------------------------------------------------------------
// Newton-Schulz GEMM: C = alpha*(A @ B) + beta*I, per batch 256^3.
// split-bf16 (x = hi + lo): 3 WMMAs per (K-step, J-tile) for ~f32 accuracy.
// 4x J-blocked: the A hi/lo fragments are built once per K-step and feed
// 12 WMMAs, halving per-WMMA load/convert overhead.
// ---------------------------------------------------------------------------
__global__ void k_ns_gemm4(const float* __restrict__ Abase, const float* __restrict__ Bbase,
                           float* __restrict__ Cbase, float alpha, float beta) {
  const int lane = threadIdx.x;
  const int mt = blockIdx.x, nq = blockIdx.y, b = blockIdx.z;
  const float* A = Abase + (size_t)b * FEAT2;
  const float* Bm = Bbase + (size_t)b * FEAT2;
  float* C = Cbase + (size_t)b * FEAT2;
  const int half = lane >> 4, l15 = lane & 15;
  const int row = mt * 16 + l15;
  const int colbase = nq * 64;
  v8f acc[4] = {{}, {}, {}, {}};
  for (int k0 = 0; k0 < FEAT; k0 += 32) {
    v16u ah, al;
    #pragma unroll
    for (int j = 0; j < 8; ++j) {
      float x = A[(size_t)row * FEAT + k0 + half * 16 + j];
      u16 h = f2bf(x); ah[j] = h; al[j] = f2bf(x - bf2f(h));
      float y = A[(size_t)row * FEAT + k0 + half * 16 + 8 + j];
      u16 h2 = f2bf(y); ah[8 + j] = h2; al[8 + j] = f2bf(y - bf2f(h2));
    }
    #pragma unroll
    for (int jt = 0; jt < 4; ++jt) {
      const int col = colbase + jt * 16 + l15;
      v16u bh, bl;
      #pragma unroll
      for (int j = 0; j < 16; ++j) {
        float x = Bm[(size_t)(k0 + half * 16 + j) * FEAT + col];
        u16 h = f2bf(x); bh[j] = h; bl[j] = f2bf(x - bf2f(h));
      }
      acc[jt] = wmma_bf16(ah, bh, acc[jt]);
      acc[jt] = wmma_bf16(ah, bl, acc[jt]);
      acc[jt] = wmma_bf16(al, bh, acc[jt]);
    }
  }
  #pragma unroll
  for (int jt = 0; jt < 4; ++jt) {
    #pragma unroll
    for (int i = 0; i < 8; ++i) {
      int r = mt * 16 + half * 8 + i;
      int c = colbase + jt * 16 + l15;
      C[(size_t)r * FEAT + c] = alpha * acc[jt][i] + ((r == c) ? beta : 0.0f);
    }
  }
}

// ---------------------------------------------------------------------------
// FC: out[b,o] = ss[b] * sum_k Y[b,k]*Wfc[o,k] + bfc[o]. HBM-bound (67 MB of
// Wfc); stream Wfc through LDS with async double-buffering:
//   global_load_async_to_lds_b128 (ASYNCcnt) + s_wait_asynccnt.
// ---------------------------------------------------------------------------
__global__ void k_fc(const float* __restrict__ Yf, const float* __restrict__ ss,
                     const float* __restrict__ Wfc, const float* __restrict__ bfc,
                     float* __restrict__ outpre) {
  __shared__ __align__(16) float wbuf[2][FC_CH];
  __shared__ float red[256];
  const int o = blockIdx.x, b = blockIdx.y, t = threadIdx.x;
  const float* w = Wfc + (size_t)o * FEAT2;
  const float* v = Yf + (size_t)b * FEAT2;
  // stage chunk 0: each lane copies 16B -> 256 threads * 16B = FC_CH floats
  {
    unsigned lds = (unsigned)(uintptr_t)&wbuf[0][t * 4];
    const float* g = w + t * 4;
    asm volatile("global_load_async_to_lds_b128 %0, %1, off" :: "v"(lds), "v"(g) : "memory");
  }
  float s = 0.f;
  for (int k0 = 0; k0 < FEAT2; k0 += FC_CH) {
    const int cur = (k0 / FC_CH) & 1;
    if (k0 + FC_CH < FEAT2) {
      unsigned lds = (unsigned)(uintptr_t)&wbuf[cur ^ 1][t * 4];
      const float* g = w + k0 + FC_CH + t * 4;
      asm volatile("global_load_async_to_lds_b128 %0, %1, off" :: "v"(lds), "v"(g) : "memory");
      asm volatile("s_wait_asynccnt 0x1" ::: "memory");   // current chunk landed
    } else {
      asm volatile("s_wait_asynccnt 0x0" ::: "memory");   // last chunk landed
    }
    __syncthreads();
    #pragma unroll
    for (int i = 0; i < FC_CH; i += 256) s += v[k0 + i + t] * wbuf[cur][i + t];
    __syncthreads();   // all waves done with `cur` before it is overwritten
  }
  red[t] = s; __syncthreads();
  for (int wd = 128; wd > 0; wd >>= 1) { if (t < wd) red[t] += red[t + wd]; __syncthreads(); }
  if (t == 0) outpre[(size_t)b * OUTD + o] = ss[b] * red[0] + bfc[o];
}

__global__ void k_l2(const float* __restrict__ pre, float* __restrict__ out) {
  __shared__ float red[256];
  int b = blockIdx.x, t = threadIdx.x;
  float v = pre[(size_t)b * OUTD + t];
  red[t] = v * v; __syncthreads();
  for (int w = 128; w > 0; w >>= 1) { if (t < w) red[t] += red[t + w]; __syncthreads(); }
  float norm = fmaxf(sqrtf(red[0]), 1e-12f);
  out[(size_t)b * OUTD + t] = v / norm;
}

// ---------------------------------------------------------------------------
// Host driver (graph-capture safe: only async launches on `stream`)
// Workspace budget ~200 MB.
// ---------------------------------------------------------------------------
extern "C" void kernel_launch(void* const* d_in, const int* in_sizes, int n_in,
                              void* d_out, int out_size, void* d_ws, size_t ws_size,
                              hipStream_t stream) {
  (void)in_sizes; (void)n_in; (void)out_size; (void)ws_size;
  const float* points = (const float*)d_in[0];
  const float* W[5]; const float* G[5]; const float* Bb[5];
  for (int i = 0; i < 5; ++i) {
    W[i]  = (const float*)d_in[1 + 3 * i];
    G[i]  = (const float*)d_in[2 + 3 * i];
    Bb[i] = (const float*)d_in[3 + 3 * i];
  }
  const float* Wfc = (const float*)d_in[16];
  const float* bfc = (const float*)d_in[17];

  char* p = (char*)d_ws;
  auto alloc = [&](size_t bytes) -> char* {
    char* r = p; p += (bytes + 255) & ~(size_t)255; return r;
  };
  u16* bufP   = (u16*)alloc((size_t)MROWS * 256 * 2);
  u16* bufQ   = (u16*)alloc((size_t)MROWS * 256 * 2);
  u16* X0p    = (u16*)alloc((size_t)MROWS * 32 * 2);
  u16* Wb[5];
  Wb[0] = (u16*)alloc(64 * 32 * 2);
  Wb[1] = (u16*)alloc(64 * 64 * 2);
  Wb[2] = (u16*)alloc(64 * 64 * 2);
  Wb[3] = (u16*)alloc(128 * 64 * 2);
  Wb[4] = (u16*)alloc(256 * 128 * 2);
  float* psum   = (float*)alloc((size_t)NCHUNK * 256 * 4);
  float* psumsq = (float*)alloc((size_t)NCHUNK * 256 * 4);
  float* sumC   = (float*)alloc(256 * 4);
  float* sumsqC = (float*)alloc(256 * 4);
  float* scaleC = (float*)alloc(256 * 4);
  float* shiftC = (float*)alloc(256 * 4);
  float* fsumC  = (float*)alloc(256 * 4);
  float* dump   = (float*)alloc(256 * 4);
  float* cov    = (float*)alloc((size_t)NB * FEAT2 * 4);
  float* nsY    = (float*)alloc((size_t)NB * FEAT2 * 4);
  float* nsZ    = (float*)alloc((size_t)NB * FEAT2 * 4);
  float* nsT    = (float*)alloc((size_t)NB * FEAT2 * 4);
  float* nsY2   = (float*)alloc((size_t)NB * FEAT2 * 4);
  float* nsZ2   = (float*)alloc((size_t)NB * FEAT2 * 4);
  float* rsArr  = (float*)alloc(NB * 4);
  float* ssArr  = (float*)alloc(NB * 4);
  float* outpre = (float*)alloc((size_t)NB * OUTD * 4);

  // --- prep ---
  k_prep_x0<<<(MROWS + 255) / 256, 256, 0, stream>>>(points, X0p);
  k_pad_w0<<<(64 * 32 + 255) / 256, 256, 0, stream>>>(W[0], Wb[0]);
  const int wsz[5] = {0, 64 * 64, 64 * 64, 128 * 64, 256 * 128};
  for (int l = 1; l < 5; ++l)
    k_cvt_bf16<<<(wsz[l] + 255) / 256, 256, 0, stream>>>(W[l], Wb[l], wsz[l]);

  // --- MLP layers (GEMM -> BN stats -> fused BN+ReLU) ---
  const int Kin[5]  = {32, 64, 64, 64, 128};
  const int Cout[5] = {64, 64, 64, 128, 256};
  for (int l = 0; l < 5; ++l) {
    const u16* Ain = (l == 0) ? X0p : bufP;
    dim3 gg(MROWS / 16, Cout[l] / 64);
    k_gemm_act4<<<gg, 32, 0, stream>>>(Ain, Kin[l], Wb[l], Kin[l], bufQ, Cout[l], Kin[l]);
    k_colstats<<<NCHUNK, Cout[l], 0, stream>>>(bufQ, Cout[l], psum, psumsq);
    k_colred<<<1, Cout[l], 0, stream>>>(psum, psumsq, Cout[l], NCHUNK, sumC, sumsqC);
    k_bnfin<<<1, Cout[l], 0, stream>>>(sumC, sumsqC, G[l], Bb[l], scaleC, shiftC);
    k_bn_apply<<<4096, 256, 0, stream>>>(bufQ, bufP, scaleC, shiftC, Cout[l],
                                         MROWS * Cout[l]);
  }

  // --- center features, covariance ---
  k_colstats<<<NCHUNK, 256, 0, stream>>>(bufP, 256, psum, psumsq);
  k_colred<<<1, 256, 0, stream>>>(psum, psumsq, 256, NCHUNK, fsumC, dump);
  k_center<<<4096, 256, 0, stream>>>(bufP, fsumC, bufQ, MROWS * FEAT);
  dim3 gc(16, 4, NB);
  k_cov4<<<gc, 32, 0, stream>>>(bufQ, cov);

  // --- Newton-Schulz matrix square root ---
  k_frob<<<NB, 256, 0, stream>>>(cov, rsArr, ssArr);
  k_ns_init<<<(NB * FEAT2) / 256, 256, 0, stream>>>(cov, rsArr, nsY, nsZ);
  float *Yc = nsY, *Zc = nsZ, *Yn = nsY2, *Zn = nsZ2;
  dim3 g3(16, 4, NB);
  for (int it = 0; it < NS_ITERS; ++it) {
    k_ns_gemm4<<<g3, 32, 0, stream>>>(Zc, Yc, nsT, -0.5f, 1.5f);  // T = 1.5I - 0.5*Z@Y
    k_ns_gemm4<<<g3, 32, 0, stream>>>(Yc, nsT, Yn, 1.0f, 0.0f);   // Y' = Y@T
    k_ns_gemm4<<<g3, 32, 0, stream>>>(nsT, Zc, Zn, 1.0f, 0.0f);   // Z' = T@Z
    float* t0 = Yc; Yc = Yn; Yn = t0;
    float* t1 = Zc; Zc = Zn; Zn = t1;
  }

  // --- FC + L2 normalize ---
  dim3 gf(OUTD, NB);
  k_fc<<<gf, 256, 0, stream>>>(Yc, ssArr, Wfc, bfc, outpre);
  k_l2<<<NB, 256, 0, stream>>>(outpre, (float*)d_out);
}
